// SGCN_24550033064283
// MI455X (gfx1250) — matrix operand order repaired and checked by reading, not compiled, not run on hardware
//
#include <hip/hip_runtime.h>

// Problem constants (match reference)
#define B_  64
#define N_  2048
#define E_  16384
#define F_  32
#define H_  64
#define K_  8
#define C_  16
#define G_  64
#define CK_ 128   // C_*K_
#define ZWP 40    // padded LDS pitch (elements) for the 128x32 zw tile: 20*q mod 64 is a permutation
#define XLP (N_ + 8)  // padded pitch for x panel in LDS: 1028 dwords -> bank offset 4*q (permutation)

#define AS1 __attribute__((address_space(1)))
#define AS3 __attribute__((address_space(3)))

#if defined(__has_builtin)
#if __has_builtin(__builtin_amdgcn_global_load_async_to_lds_b128)
#define HAVE_ASYNC_LDS 1
#endif
#endif

typedef __attribute__((ext_vector_type(16))) __bf16 bf16x16;
typedef __attribute__((ext_vector_type(8)))  __bf16 bf16x8;
typedef __attribute__((ext_vector_type(8)))  float  f32x8;
typedef __attribute__((ext_vector_type(4)))  unsigned int u32x4;
typedef int v4i_gcc __attribute__((vector_size(16)));   // matches builtin's expected pointee

// Build a 16-element bf16 fragment from two 16-byte loads (works for global and LDS pointers).
static __device__ inline bf16x16 frag_from2(const __bf16* p0, const __bf16* p1) {
  bf16x8 lo = *(const bf16x8*)p0;
  bf16x8 hi = *(const bf16x8*)p1;
  bf16x16 r;
#pragma unroll
  for (int i = 0; i < 8; ++i) { r[i] = lo[i]; r[8 + i] = hi[i]; }
  return r;
}

// Packed bf16 multiply: one V_PK_MUL_BF16 per dword pair (CDNA5 VOP3P).
static __device__ inline unsigned int pkmul_bf16(unsigned int a, unsigned int b) {
  unsigned int d;
  asm volatile("v_pk_mul_bf16 %0, %1, %2" : "=v"(d) : "v"(a), "v"(b));
  return d;
}

static __device__ inline u32x4 pkmul4(u32x4 a, u32x4 b) {
  u32x4 d;
#pragma unroll
  for (int i = 0; i < 4; ++i) d[i] = pkmul_bf16(a[i], b[i]);
  return d;
}

// Async global->LDS 16-byte copy (ASYNCcnt path) with safe fallback.
static __device__ inline void lds_copy16(const __bf16* g, __bf16* l) {
#if HAVE_ASYNC_LDS
  __builtin_amdgcn_global_load_async_to_lds_b128((AS1 v4i_gcc*)g, (AS3 v4i_gcc*)l, 0, 0);
#else
  *(bf16x8*)l = *(const bf16x8*)g;
#endif
}

static __device__ inline void lds_copy_join() {
#if HAVE_ASYNC_LDS
#if __has_builtin(__builtin_amdgcn_s_wait_asynccnt)
  __builtin_amdgcn_s_wait_asynccnt(0);
#else
  asm volatile("s_wait_asynccnt 0x0" ::: "memory");
#endif
#endif
}

// ---------------- conversion kernels ----------------

__global__ void cvt_f32_bf16(const float* __restrict__ s, __bf16* __restrict__ d, long n) {
  long i = (long)blockIdx.x * blockDim.x + threadIdx.x;
  if (i < n) d[i] = (__bf16)s[i];
}

// x[B][N][C] fp32 -> xT[B][C][N] bf16 (so GEMM1's B operand is K-contiguous per lane)
__global__ void xpose_x(const float* __restrict__ x, __bf16* __restrict__ xT) {
  long i = (long)blockIdx.x * blockDim.x + threadIdx.x;
  if (i >= (long)B_ * N_ * C_) return;
  int c = (int)(i % C_);
  long t = i / C_;
  int n = (int)(t % N_);
  int b = (int)(t / N_);
  xT[((long)b * C_ + c) * N_ + n] = (__bf16)x[i];
}

// ---------------- edge MLP: w[e,k] = (relu(ef@w1^T+b1))@w2^T+b2, stored transposed bf16 ----------------

__global__ __launch_bounds__(64) void edge_mlp(const float* __restrict__ ef,
                                               const float* __restrict__ w1,
                                               const float* __restrict__ b1,
                                               const float* __restrict__ w2,
                                               const float* __restrict__ b2,
                                               __bf16* __restrict__ wT) {
  __shared__ float sw1[H_ * F_];
  __shared__ float sb1[H_];
  __shared__ float sw2[K_ * H_];
  __shared__ float sb2[K_];
  int t = threadIdx.x;
  for (int i = t; i < H_ * F_; i += 64) sw1[i] = w1[i];
  for (int i = t; i < H_;      i += 64) sb1[i] = b1[i];
  for (int i = t; i < K_ * H_; i += 64) sw2[i] = w2[i];
  if (t < K_) sb2[t] = b2[t];
  __syncthreads();

  int e = blockIdx.x * 64 + t;
  float r[F_];
#pragma unroll
  for (int f = 0; f < F_; ++f) r[f] = ef[(long)e * F_ + f];
  float h[H_];
  for (int j = 0; j < H_; ++j) {
    float acc = sb1[j];
#pragma unroll
    for (int f = 0; f < F_; ++f) acc += r[f] * sw1[j * F_ + f];
    h[j] = acc > 0.f ? acc : 0.f;
  }
#pragma unroll
  for (int k = 0; k < K_; ++k) {
    float acc = sb2[k];
    for (int j = 0; j < H_; ++j) acc += h[j] * sw2[k * H_ + j];
    wT[(long)k * E_ + e] = (__bf16)acc;
  }
}

// ---------------- GEMM1: zT[b][c][e] = sum_n rinc[e,n] * x[b,n,c]  (bf16 WMMA) ----------------
// grid: (E/128, B), block 256 (8 waves). The shared 16x2048 x-panel (64 KB) is staged into LDS
// once per block with async global->LDS copies; all 8 waves serve B-fragments from LDS.

__global__ __launch_bounds__(256) void gemm1_z(const __bf16* __restrict__ rincb,
                                               const __bf16* __restrict__ xT,
                                               __bf16* __restrict__ zT) {
  __shared__ __align__(16) __bf16 xlds[C_][XLP];   // ~64.25 KB (padded rows)
  int tid  = threadIdx.x;
  int wave = tid >> 5, lane = tid & 31;
  int half = lane >> 4, q = lane & 15;
  int b  = blockIdx.y;
  int e0 = blockIdx.x * 128 + wave * 16;

  // Stage the x^T panel for this batch: 4096 x 16B chunks, 16 per thread (async).
  const __bf16* xsrc = xT + (size_t)b * C_ * N_;
  for (int i = tid; i < (C_ * N_) / 8; i += 256) {
    int c   = i >> 8;            // 256 chunks of 8 elems per row
    int off = (i & 255) * 8;
    lds_copy16(xsrc + (size_t)c * N_ + off, &xlds[c][off]);
  }
  lds_copy_join();
  __syncthreads();

  const __bf16* arow = rincb + (size_t)(e0 + q) * N_;  // A row m = q
  const __bf16* brow = &xlds[q][16 * half];            // B col n = q served from LDS

  f32x8 acc = {};
  for (int k0 = 0; k0 < N_; k0 += 32) {
    bf16x16 af = frag_from2(arow + k0 + 16 * half, arow + k0 + 16 * half + 8);
    bf16x16 bf = frag_from2(brow + k0,            brow + k0 + 8);
    acc = __builtin_amdgcn_wmma_f32_16x16x32_bf16(false, af, false, bf, (short)0, acc, false, false);
  }
  // D: lane column n=q (c index), row m = r + 8*half -> zT[b][q][e0 + r + 8*half]
  __bf16* zp = zT + ((size_t)b * C_ + q) * E_ + e0 + 8 * half;
#pragma unroll
  for (int r = 0; r < 8; ++r) zp[r] = (__bf16)acc[r];
}

// ---------------- GEMM2 + fused output projection ----------------
// y[n, ck] = sum_e linc[n,e] * (w[e,k] * z[b,e,c]),  out[n,g] = relu(sum_ck y*gcw[g,ck] + gcb[g])
// grid: (N/128, B), block 256 (8 waves). The zw B-tile [128 ck x 32 e] is built cooperatively
// in LDS (double-buffered, V_PK_MUL_BF16) so all 8 waves share it.

__global__ __launch_bounds__(256) void gemm23(const __bf16* __restrict__ lincb,
                                              const __bf16* __restrict__ zT,
                                              const __bf16* __restrict__ wT,
                                              const __bf16* __restrict__ gcwb,
                                              const float* __restrict__ gcb,
                                              float* __restrict__ out) {
  __shared__ __align__(16) __bf16 zwlds[2][CK_][ZWP]; // 2 x 10 KB staged B tiles
  __shared__ __align__(16) __bf16 ylds[8][16][CK_];   // 32 KB y-tile reshaping buffer
  int tid  = threadIdx.x;
  int wave = tid >> 5, lane = tid & 31;
  int half = lane >> 4, q = lane & 15;
  int b  = blockIdx.y;
  int n0 = blockIdx.x * 128 + wave * 16;

  const __bf16* arow = lincb + (size_t)(n0 + q) * E_;

  // Staging role: thread -> (ck column sc, e-half sh); builds zw[sc][16*sh .. 16*sh+15]
  int sc = tid >> 1;          // 0..127
  int sh = tid & 1;           // 0..1
  int scc = sc >> 3, sck = sc & 7;
  const __bf16* zsrc = zT + ((size_t)b * C_ + scc) * E_ + 16 * sh;
  const __bf16* wsrc = wT + (size_t)sck * E_ + 16 * sh;
  __bf16* zwdst = &zwlds[0][sc][16 * sh];
  const size_t bufstep = (size_t)CK_ * ZWP;  // elements between buffers

  f32x8 acc[8];
#pragma unroll
  for (int t = 0; t < 8; ++t) acc[t] = (f32x8){};

  // Prologue: stage chunk 0 into buffer 0.
  {
    u32x4 z0 = *(const u32x4*)(zsrc);
    u32x4 z1 = *(const u32x4*)(zsrc + 8);
    u32x4 w0 = *(const u32x4*)(wsrc);
    u32x4 w1 = *(const u32x4*)(wsrc + 8);
    *(u32x4*)(zwdst)     = pkmul4(z0, w0);
    *(u32x4*)(zwdst + 8) = pkmul4(z1, w1);
  }

  const int nch = E_ / 32;
  int buf = 0;
  for (int i = 0; i < nch; ++i) {
    __syncthreads();   // staged tile [buf] ready; previous reads of [buf^1] complete
    int e0 = i * 32;
    if (i + 1 < nch) {  // stage next chunk into the other buffer
      int e1 = e0 + 32;
      u32x4 z0 = *(const u32x4*)(zsrc + e1);
      u32x4 z1 = *(const u32x4*)(zsrc + e1 + 8);
      u32x4 w0 = *(const u32x4*)(wsrc + e1);
      u32x4 w1 = *(const u32x4*)(wsrc + e1 + 8);
      __bf16* d = zwdst + (buf ^ 1) * bufstep;
      *(u32x4*)(d)     = pkmul4(z0, w0);
      *(u32x4*)(d + 8) = pkmul4(z1, w1);
    }
    if (i + 16 < nch) __builtin_prefetch(arow + e0 + 512, 0, 1);

    bf16x16 af = frag_from2(arow + e0 + 16 * half, arow + e0 + 16 * half + 8);
#pragma unroll
    for (int t = 0; t < 8; ++t) {
      const __bf16* bp = &zwlds[buf][t * 16 + q][16 * half];
      bf16x16 bfr = frag_from2(bp, bp + 8);
      acc[t] = __builtin_amdgcn_wmma_f32_16x16x32_bf16(false, af, false, bfr, (short)0, acc[t], false, false);
    }
    buf ^= 1;
  }

  // Stage y tile (bf16) through LDS to reshape D-fragments into A-fragments.
  __syncthreads();
#pragma unroll
  for (int t = 0; t < 8; ++t)
#pragma unroll
    for (int r = 0; r < 8; ++r)
      ylds[wave][r + 8 * half][t * 16 + q] = (__bf16)acc[t][r];
  __syncthreads();

  const __bf16* yrow = &ylds[wave][q][0];   // A row m = q, K = 128
  bf16x16 afr[4];
#pragma unroll
  for (int kc = 0; kc < 4; ++kc)
    afr[kc] = frag_from2(yrow + kc * 32 + 16 * half, yrow + kc * 32 + 16 * half + 8);

#pragma unroll
  for (int gt = 0; gt < 4; ++gt) {
    const __bf16* gp = gcwb + (size_t)(gt * 16 + q) * CK_ + 16 * half;  // B col n = g
    f32x8 a3 = {};
#pragma unroll
    for (int kc = 0; kc < 4; ++kc) {
      bf16x16 bfr = frag_from2(gp + kc * 32, gp + kc * 32 + 8);
      a3 = __builtin_amdgcn_wmma_f32_16x16x32_bf16(false, afr[kc], false, bfr, (short)0, a3, false, false);
    }
    float bias = gcb[gt * 16 + q];
    float* op = out + ((size_t)b * N_ + n0 + 8 * half) * G_ + gt * 16 + q;
#pragma unroll
    for (int r = 0; r < 8; ++r) {
      float v = a3[r] + bias;
      op[(size_t)r * G_] = v > 0.f ? v : 0.f;
    }
  }
}

// ---------------- host launcher ----------------

extern "C" void kernel_launch(void* const* d_in, const int* in_sizes, int n_in,
                              void* d_out, int out_size, void* d_ws, size_t ws_size,
                              hipStream_t stream) {
  const float* x    = (const float*)d_in[0];
  const float* linc = (const float*)d_in[1];
  const float* rinc = (const float*)d_in[2];
  const float* ef   = (const float*)d_in[3];
  const float* w1   = (const float*)d_in[4];
  const float* b1   = (const float*)d_in[5];
  const float* w2   = (const float*)d_in[6];
  const float* b2   = (const float*)d_in[7];
  const float* gcw  = (const float*)d_in[8];
  const float* gcb  = (const float*)d_in[9];
  float* out = (float*)d_out;

  char* ws = (char*)d_ws;
  // Workspace layout (all offsets 4KiB-aligned); total ~164 MiB
  __bf16* rincb = (__bf16*)(ws);                       // E*N bf16   = 64 MiB
  __bf16* lincb = (__bf16*)(ws + (size_t)67108864);    // N*E bf16   = 64 MiB
  __bf16* xT    = (__bf16*)(ws + (size_t)134217728);   // B*C*N bf16 =  4 MiB
  __bf16* zT    = (__bf16*)(ws + (size_t)138412032);   // B*C*E bf16 = 32 MiB
  __bf16* wT    = (__bf16*)(ws + (size_t)171966464);   // K*E bf16   = 256 KiB
  __bf16* gcwb  = (__bf16*)(ws + (size_t)172228608);   // G*CK bf16  =  16 KiB

  const long nInc = (long)E_ * N_;
  cvt_f32_bf16<<<(unsigned)((nInc + 255) / 256), 256, 0, stream>>>(rinc, rincb, nInc);
  cvt_f32_bf16<<<(unsigned)((nInc + 255) / 256), 256, 0, stream>>>(linc, lincb, nInc);
  cvt_f32_bf16<<<(G_ * CK_ + 255) / 256, 256, 0, stream>>>(gcw, gcwb, (long)G_ * CK_);

  const long nX = (long)B_ * N_ * C_;
  xpose_x<<<(unsigned)((nX + 255) / 256), 256, 0, stream>>>(x, xT);

  edge_mlp<<<E_ / 64, 64, 0, stream>>>(ef, w1, b1, w2, b2, wT);

  gemm1_z<<<dim3(E_ / 128, B_), 256, 0, stream>>>(rincb, xT, zT);

  gemm23<<<dim3(N_ / 128, B_), 256, 0, stream>>>(lincb, zT, wT, gcwb, gcb, out);
}
